// Attention_69234872811552
// MI455X (gfx1250) — matrix operand order repaired and checked by reading, not compiled, not run on hardware
//
#include <hip/hip_runtime.h>
#include <hip/hip_bf16.h>

#define HID 1024
#define HEADS 16
#define DPH 64
#define BATCH 2
#define FQ 2048
#define TK 2048

#if defined(__has_builtin)
#define HAS_BUILTIN(x) __has_builtin(x)
#else
#define HAS_BUILTIN(x) 0
#endif

typedef __bf16 bf16;
typedef __attribute__((ext_vector_type(16))) __bf16 v16bf;
typedef __attribute__((ext_vector_type(8)))  __bf16 v8bf;
typedef __attribute__((ext_vector_type(4)))  __bf16 v4bf;
typedef __attribute__((ext_vector_type(8)))  float  v8f;
typedef __attribute__((ext_vector_type(4)))  float  v4f;
typedef __attribute__((ext_vector_type(4)))  unsigned int u32x4;
typedef __attribute__((ext_vector_type(8)))  int i32x8;
typedef __attribute__((ext_vector_type(4)))  int i32x4;

// LDS layout (attn_kernel, dynamic shared only => offsets start at 0)
#define QS_ELE   0
#define KS_ELE   (128 * 64)
#define VS_ELE   (KS_ELE + 128 * 64)
#define QS_BYTE  0u
#define KS_BYTE  16384u

union AFrag { v16bf v; v8bf h[2]; };

// A fragment (16x32 bf16): lane = M row (lane&15); lanes<16 get K {0..7,16..23},
// lanes>=16 get K {8..15,24..31}.  base points at row 0, k 0; ldk in elements.
__device__ __forceinline__ v16bf lds_load_a(const bf16* base, int ldk, int lane) {
  const bf16* p = base + (lane & 15) * ldk + ((lane >> 4) << 3);
  AFrag u;
  u.h[0] = *(const v8bf*)p;
  u.h[1] = *(const v8bf*)(p + 16);
  return u.v;
}

// B fragment (32x16 bf16) from a [N][K]-transposed LDS tile: lane = N col (lane&15);
// lanes 0-15 hold K {0..7,16..23}, lanes 16-31 hold K {8..15,24..31} (matches A).
__device__ __forceinline__ v16bf lds_load_b(const bf16* base, int ldk, int lane) {
  const bf16* p = base + (lane & 15) * ldk + ((lane >> 4) << 3);
  AFrag u;
  u.h[0] = *(const v8bf*)p;
  u.h[1] = *(const v8bf*)(p + 16);
  return u.v;
}

__device__ __forceinline__ v8f wmma_bf16(v16bf a, v16bf b, v8f c) {
  return __builtin_amdgcn_wmma_f32_16x16x32_bf16(false, a, false, b, (short)0, c, false, false);
}

// Per-lane async global->LDS copy of 16 bytes (ASYNCcnt path, no VGPR round trip).
__device__ __forceinline__ void async_copy_b128(unsigned lds_byte_off, const void* gaddr) {
#if defined(__AMDGCN__)
  asm volatile("global_load_async_to_lds_b128 %0, %1, off"
               :: "v"(lds_byte_off), "v"(gaddr) : "memory");
#endif
}

__device__ __forceinline__ void wait_async0() {
#if defined(__AMDGCN__)
  asm volatile("s_wait_asynccnt 0x0" ::: "memory");
#endif
}

// ---------------------------------------------------------------------------
// QKV projection: Y[m][n] = (bf16)(scale * sum_k X[m][k] * W[k][n])
// X: f32 [4096][1024], W: f32 [1024][1024], Y: bf16 [4096][1024]
// ---------------------------------------------------------------------------
__global__ __launch_bounds__(256) void proj_gemm(const float* __restrict__ X,
                                                 const float* __restrict__ W,
                                                 bf16* __restrict__ Y,
                                                 float scale) {
  __shared__ bf16 a_lds[128 * 64];   // [m][k]
  __shared__ bf16 b_lds[128 * 64];   // [n][k] (transposed)
  const int tid  = threadIdx.x;
  const int lane = tid & 31;
  const int wave = tid >> 5;
  const int m0 = blockIdx.x * 128;
  const int n0 = blockIdx.y * 128;
  const int wr = (wave >> 1) * 32;
  const int wc = (wave & 1) * 64;

  v8f acc[2][4];
#pragma unroll
  for (int mt = 0; mt < 2; ++mt)
#pragma unroll
    for (int nt = 0; nt < 4; ++nt)
#pragma unroll
      for (int e = 0; e < 8; ++e) acc[mt][nt][e] = 0.0f;

  for (int k0 = 0; k0 < HID; k0 += 64) {
    __syncthreads();
#pragma unroll
    for (int c = tid; c < 2048; c += 256) {
      int row = c >> 4, kb = (c & 15) << 2;
      v4f xv = *(const v4f*)(X + (size_t)(m0 + row) * HID + k0 + kb);
      v4bf bv;
      bv[0] = (bf16)xv[0]; bv[1] = (bf16)xv[1]; bv[2] = (bf16)xv[2]; bv[3] = (bf16)xv[3];
      *(v4bf*)(a_lds + row * 64 + kb) = bv;
    }
#pragma unroll
    for (int c = tid; c < 2048; c += 256) {
      int kr = c >> 5, nb = (c & 31) << 2;
      v4f wv = *(const v4f*)(W + (size_t)(k0 + kr) * HID + n0 + nb);
      b_lds[(nb + 0) * 64 + kr] = (bf16)wv[0];
      b_lds[(nb + 1) * 64 + kr] = (bf16)wv[1];
      b_lds[(nb + 2) * 64 + kr] = (bf16)wv[2];
      b_lds[(nb + 3) * 64 + kr] = (bf16)wv[3];
    }
    if (k0 + 64 < HID) __builtin_prefetch(X + (size_t)(m0 + (tid >> 1)) * HID + k0 + 64, 0, 0);
    __syncthreads();
#pragma unroll
    for (int ks = 0; ks < 64; ks += 32) {
      v16bf af[2], bfv[4];
      af[0] = lds_load_a(a_lds + (wr +  0) * 64 + ks, 64, lane);
      af[1] = lds_load_a(a_lds + (wr + 16) * 64 + ks, 64, lane);
#pragma unroll
      for (int nt = 0; nt < 4; ++nt)
        bfv[nt] = lds_load_b(b_lds + (wc + nt * 16) * 64 + ks, 64, lane);
#pragma unroll
      for (int mt = 0; mt < 2; ++mt)
#pragma unroll
        for (int nt = 0; nt < 4; ++nt)
          acc[mt][nt] = wmma_bf16(af[mt], bfv[nt], acc[mt][nt]);
    }
  }
  const int rbase = m0 + wr + 8 * (lane >> 4);
  const int cbase = n0 + wc + (lane & 15);
#pragma unroll
  for (int mt = 0; mt < 2; ++mt)
#pragma unroll
    for (int nt = 0; nt < 4; ++nt)
#pragma unroll
      for (int e = 0; e < 8; ++e)
        Y[(size_t)(rbase + mt * 16 + e) * HID + cbase + nt * 16] =
            (bf16)(acc[mt][nt][e] * scale);
}

// ---------------------------------------------------------------------------
// Flash attention: one (b, head, 128-row F tile) per workgroup.
// ---------------------------------------------------------------------------
__global__ __launch_bounds__(256) void attn_kernel(const bf16* __restrict__ q,
                                                   const bf16* __restrict__ k,
                                                   const bf16* __restrict__ v,
                                                   const float* __restrict__ bias,
                                                   bf16* __restrict__ attn) {
  extern __shared__ char smem[];
  bf16*  qs    = (bf16*)smem;                    // [128 f][64 d]
  bf16*  ks    = qs + KS_ELE - QS_ELE + QS_ELE;  // [128 t][64 d]
  float* row_m;
  ks = ((bf16*)smem) + KS_ELE;
  bf16*  vs    = ((bf16*)smem) + VS_ELE;         // [64 d][128 t] (transposed)
  float* s_lds = (float*)(vs + 64 * 128);        // [128 f][128 t]
  bf16*  p_lds = (bf16*)(s_lds + 128 * 128);     // [128 f][128 t]
  row_m        = (float*)(p_lds + 128 * 128);    // [128]
  float* row_l = row_m + 128;
  float* row_c = row_l + 128;

  const int tid  = threadIdx.x;
  const int lane = tid & 31;
  const int wave = tid >> 5;
  const int b  = blockIdx.z;
  const int n  = blockIdx.y;
  const int f0 = blockIdx.x * 128;
  const size_t qoff = (size_t)b * FQ * HID + (size_t)n * DPH;

  // Q tile: async global->LDS (bf16 copy, no conversion needed)
#pragma unroll
  for (int c = tid; c < 1024; c += 256) {
    int r = c >> 3, db = (c & 7) << 3;
    async_copy_b128(QS_BYTE + (unsigned)((r * 64 + db) * 2),
                    (const void*)(q + qoff + (size_t)(f0 + r) * HID + db));
  }
  if (tid < 128) { row_m[tid] = -3.0e38f; row_l[tid] = 0.0f; }

  v8f oacc[4];
#pragma unroll
  for (int dt = 0; dt < 4; ++dt)
#pragma unroll
    for (int e = 0; e < 8; ++e) oacc[dt][e] = 0.0f;
  const int fw = wave * 16;

  for (int t0 = 0; t0 < TK; t0 += 128) {
    __syncthreads();
    const size_t koff = ((size_t)b * TK + t0) * HID + (size_t)n * DPH;

    // ---- K tile [128 t][64 d]: Tensor Data Mover if available, else async ----
#if HAS_BUILTIN(__builtin_amdgcn_tensor_load_to_lds)
    if (wave == 0) {
      const unsigned long long ga = (unsigned long long)(const void*)(k + koff);
      u32x4 g0;
      g0[0] = 1u;                                    // count=1, user descriptor
      g0[1] = KS_BYTE;                               // lds_addr (bytes)
      g0[2] = (unsigned)ga;                          // global_addr[31:0]
      g0[3] = (unsigned)((ga >> 32) & 0x01ffffffu)   // global_addr[56:32]
              | (2u << 30);                          // type=2 ("image")
      i32x8 g1;
      g1[0] = (int)(1u << 16);        // data_size=1 (2 bytes), wg_mask=0
      g1[1] = (int)(1024u << 16);     // tensor_dim0[15:0] = 1024 (row length)
      g1[2] = (int)(4096u << 16);     // tensor_dim0 hi=0, tensor_dim1[15:0]=4096
      g1[3] = (int)(64u << 16);       // tensor_dim1 hi=0, tile_dim0=64 (d)
      g1[4] = 128;                    // tile_dim1=128 (t), tile_dim2=0
      g1[5] = 1024;                   // tensor_dim0_stride[31:0] = 1024 elements
      g1[6] = 0;
      g1[7] = 0;
      i32x4 z4 = {0, 0, 0, 0};
#if __clang_major__ >= 23
      i32x8 z8 = {0, 0, 0, 0, 0, 0, 0, 0};
      __builtin_amdgcn_tensor_load_to_lds(g0, g1, z4, z4, z8, 0);
#else
      __builtin_amdgcn_tensor_load_to_lds(g0, g1, z4, z4, 0);
#endif
#if HAS_BUILTIN(__builtin_amdgcn_s_wait_tensorcnt)
      __builtin_amdgcn_s_wait_tensorcnt(0);
#elif defined(__AMDGCN__)
      asm volatile("s_wait_tensorcnt 0x0" ::: "memory");
#endif
    }
#else
#pragma unroll
    for (int c = tid; c < 1024; c += 256) {
      int t = c >> 3, db = (c & 7) << 3;
      async_copy_b128(KS_BYTE + (unsigned)((t * 64 + db) * 2),
                      (const void*)(k + koff + (size_t)t * HID + db));
    }
#endif

    // ---- V tile transposed [64 d][128 t] (needs per-element transpose) ----
#pragma unroll
    for (int c = tid; c < 1024; c += 256) {
      int t = c >> 3, db = (c & 7) << 3;
      v8bf vv = *(const v8bf*)(v + koff + (size_t)t * HID + db);
#pragma unroll
      for (int j = 0; j < 8; ++j) vs[(db + j) * 128 + t] = vv[j];
    }
    if (t0 + 128 < TK)
      __builtin_prefetch(k + koff + 128 * HID + (size_t)(tid >> 1) * HID, 0, 0);
    wait_async0();   // Q (first iter) + K async copies from this wave
    __syncthreads();

    // S = Q * K^T : wave computes 32 f-rows x 64 t-cols
    {
      const int sr = (wave >> 1) * 32, sc = (wave & 1) * 64;
      v8f sacc[2][4];
#pragma unroll
      for (int mt = 0; mt < 2; ++mt)
#pragma unroll
        for (int nt = 0; nt < 4; ++nt)
#pragma unroll
          for (int e = 0; e < 8; ++e) sacc[mt][nt][e] = 0.0f;
#pragma unroll
      for (int ksp = 0; ksp < 64; ksp += 32) {
        v16bf af[2];
        af[0] = lds_load_a(qs + (sr +  0) * 64 + ksp, 64, lane);
        af[1] = lds_load_a(qs + (sr + 16) * 64 + ksp, 64, lane);
#pragma unroll
        for (int nt = 0; nt < 4; ++nt) {
          v16bf bfrag = lds_load_b(ks + (sc + nt * 16) * 64 + ksp, 64, lane);
          sacc[0][nt] = wmma_bf16(af[0], bfrag, sacc[0][nt]);
          sacc[1][nt] = wmma_bf16(af[1], bfrag, sacc[1][nt]);
        }
      }
      const int rb = sr + 8 * (lane >> 4);
      const int cb = sc + (lane & 15);
#pragma unroll
      for (int mt = 0; mt < 2; ++mt)
#pragma unroll
        for (int nt = 0; nt < 4; ++nt)
#pragma unroll
          for (int e = 0; e < 8; ++e)
            s_lds[(rb + mt * 16 + e) * 128 + cb + nt * 16] = sacc[mt][nt][e];
    }
    __syncthreads();

    // online softmax per row (threads 0..127), add bias
    if (tid < 128) {
      const int r = tid;
      const float* brow = bias + ((size_t)b * FQ + f0 + r) * TK + t0;
      float* srow = s_lds + r * 128;
      float mx = -3.0e38f;
      for (int c2 = 0; c2 < 128; ++c2) {
        float x = srow[c2] + brow[c2];
        srow[c2] = x;
        mx = fmaxf(mx, x);
      }
      const float mold = row_m[r];
      const float mnew = fmaxf(mold, mx);
      const float corr = __expf(mold - mnew);
      float sum = 0.0f;
      bf16* prow = p_lds + r * 128;
      for (int c2 = 0; c2 < 128; ++c2) {
        float p = __expf(srow[c2] - mnew);
        sum += p;
        prow[c2] = (bf16)p;
      }
      row_m[r] = mnew;
      row_l[r] = row_l[r] * corr + sum;
      row_c[r] = corr;
    }
    __syncthreads();

    // rescale O and accumulate P * V (wave owns rows [fw, fw+16))
    float cf[8];
#pragma unroll
    for (int e = 0; e < 8; ++e) cf[e] = row_c[fw + e + 8 * (lane >> 4)];
#pragma unroll
    for (int dt = 0; dt < 4; ++dt)
#pragma unroll
      for (int e = 0; e < 8; ++e) oacc[dt][e] *= cf[e];
#pragma unroll
    for (int kt = 0; kt < 4; ++kt) {
      v16bf af = lds_load_a(p_lds + fw * 128 + kt * 32, 128, lane);
#pragma unroll
      for (int dt = 0; dt < 4; ++dt) {
        v16bf bfrag = lds_load_b(vs + (dt * 16) * 128 + kt * 32, 128, lane);
        oacc[dt] = wmma_bf16(af, bfrag, oacc[dt]);
      }
    }
  }
  __syncthreads();

  float rinv[8];
#pragma unroll
  for (int e = 0; e < 8; ++e) rinv[e] = 1.0f / row_l[fw + e + 8 * (lane >> 4)];
  const int col = lane & 15;
#pragma unroll
  for (int dt = 0; dt < 4; ++dt)
#pragma unroll
    for (int e = 0; e < 8; ++e)
      attn[((size_t)b * FQ + f0 + fw + e + 8 * (lane >> 4)) * HID +
           (size_t)n * DPH + dt * 16 + col] = (bf16)(oacc[dt][e] * rinv[e]);
}

// ---------------------------------------------------------------------------
// Output projection: out[m][h] = sum_{nd} attn[m][nd] * wo[nd][h]
// ---------------------------------------------------------------------------
__global__ __launch_bounds__(256) void out_gemm(const bf16* __restrict__ A,
                                                const float* __restrict__ W,
                                                float* __restrict__ out) {
  __shared__ bf16 a_lds[128 * 64];
  __shared__ bf16 b_lds[128 * 64];
  const int tid  = threadIdx.x;
  const int lane = tid & 31;
  const int wave = tid >> 5;
  const int m0 = blockIdx.x * 128;
  const int n0 = blockIdx.y * 128;
  const int wr = (wave >> 1) * 32;
  const int wc = (wave & 1) * 64;

  v8f acc[2][4];
#pragma unroll
  for (int mt = 0; mt < 2; ++mt)
#pragma unroll
    for (int nt = 0; nt < 4; ++nt)
#pragma unroll
      for (int e = 0; e < 8; ++e) acc[mt][nt][e] = 0.0f;

  for (int k0 = 0; k0 < HID; k0 += 64) {
    __syncthreads();
#pragma unroll
    for (int c = tid; c < 1024; c += 256) {
      int row = c >> 3, kb = (c & 7) << 3;
      *(v8bf*)(a_lds + row * 64 + kb) =
          *(const v8bf*)(A + (size_t)(m0 + row) * HID + k0 + kb);
    }
#pragma unroll
    for (int c = tid; c < 2048; c += 256) {
      int kr = c >> 5, nb = (c & 31) << 2;
      v4f wv = *(const v4f*)(W + (size_t)(k0 + kr) * HID + n0 + nb);
      b_lds[(nb + 0) * 64 + kr] = (bf16)wv[0];
      b_lds[(nb + 1) * 64 + kr] = (bf16)wv[1];
      b_lds[(nb + 2) * 64 + kr] = (bf16)wv[2];
      b_lds[(nb + 3) * 64 + kr] = (bf16)wv[3];
    }
    __syncthreads();
#pragma unroll
    for (int ks = 0; ks < 64; ks += 32) {
      v16bf af[2], bfv[4];
      af[0] = lds_load_a(a_lds + (wr +  0) * 64 + ks, 64, lane);
      af[1] = lds_load_a(a_lds + (wr + 16) * 64 + ks, 64, lane);
#pragma unroll
      for (int nt = 0; nt < 4; ++nt)
        bfv[nt] = lds_load_b(b_lds + (wc + nt * 16) * 64 + ks, 64, lane);
#pragma unroll
      for (int mt = 0; mt < 2; ++mt)
#pragma unroll
        for (int nt = 0; nt < 4; ++nt)
          acc[mt][nt] = wmma_bf16(af[mt], bfv[nt], acc[mt][nt]);
    }
  }
  const int rbase = m0 + wr + 8 * (lane >> 4);
  const int cbase = n0 + wc + (lane & 15);
#pragma unroll
  for (int mt = 0; mt < 2; ++mt)
#pragma unroll
    for (int nt = 0; nt < 4; ++nt)
#pragma unroll
      for (int e = 0; e < 8; ++e)
        out[(size_t)(rbase + mt * 16 + e) * HID + cbase + nt * 16] = acc[mt][nt][e];
}

extern "C" void kernel_launch(void* const* d_in, const int* in_sizes, int n_in,
                              void* d_out, int out_size, void* d_ws, size_t ws_size,
                              hipStream_t stream) {
  const float* query  = (const float*)d_in[0];
  const float* source = (const float*)d_in[1];
  const float* bias   = (const float*)d_in[2];
  const float* wq     = (const float*)d_in[3];
  const float* wk     = (const float*)d_in[4];
  const float* wv     = (const float*)d_in[5];
  const float* wo     = (const float*)d_in[6];
  float* out = (float*)d_out;

  const size_t per = (size_t)BATCH * FQ * HID;
  bf16* qb = (bf16*)d_ws;
  bf16* kb = qb + per;
  bf16* vb = kb + per;
  bf16* ab = vb + per;

  dim3 blk(256);
  dim3 gproj((BATCH * FQ) / 128, HID / 128);

  proj_gemm<<<gproj, blk, 0, stream>>>(query,  wq, qb, 0.125f);
  proj_gemm<<<gproj, blk, 0, stream>>>(source, wk, kb, 1.0f);
  proj_gemm<<<gproj, blk, 0, stream>>>(source, wv, vb, 1.0f);

  dim3 gattn(FQ / 128, HEADS, BATCH);
  const size_t smem_bytes =
      (size_t)(128 * 64 + 128 * 64 + 64 * 128) * sizeof(bf16) +
      (size_t)(128 * 128) * sizeof(float) +
      (size_t)(128 * 128) * sizeof(bf16) +
      (size_t)(3 * 128) * sizeof(float);
  attn_kernel<<<gattn, blk, smem_bytes, stream>>>(qb, kb, vb, bias, ab);

  out_gemm<<<gproj, blk, 0, stream>>>(ab, wo, out);
}
